// DPSA_52905407152224
// MI455X (gfx1250) — matrix operand order repaired and checked by reading, compiled, not run on hardware
//
#include <hip/hip_runtime.h>

// ---------------------------------------------------------------------------
// DPSA for MI455X (gfx1250): bf16 WMMA GEMMs + flash-attention with fused
// dual-score (semantic + gamma*salience) softmax.  Wave32, 16x16x32 bf16 WMMA.
// All WMMA fragment loads are two contiguous 8-element (2x 16B b128) reads;
// same-major tiles are staged with global_load_async_to_lds_b128 (ASYNCcnt).
// ---------------------------------------------------------------------------

typedef __attribute__((ext_vector_type(16))) __bf16 v16bf;
typedef __attribute__((ext_vector_type(8)))  __bf16 v8bf;
typedef __attribute__((ext_vector_type(8)))  float  v8f;

#define DEV static __device__ __forceinline__

constexpr int Bb = 2, Ss = 2048, Dd = 1024, Hh = 16, DKk = 64;

DEV __bf16 f2bf(float f) {
  unsigned u = __builtin_bit_cast(unsigned, f);
  unsigned r = u + 0x7FFFu + ((u >> 16) & 1u);   // round-to-nearest-even
  unsigned short h = (unsigned short)(r >> 16);
  return __builtin_bit_cast(__bf16, h);
}

DEV v8f vzero() {
  v8f t;
#pragma unroll
  for (int i = 0; i < 8; ++i) t[i] = 0.0f;
  return t;
}

DEV v8f wmma_bf16(v16bf a, v16bf b, v8f c) {
  // D = A(16x32) * B(32x16) + C  -> v_wmma_f32_16x16x32_bf16
  return __builtin_amdgcn_wmma_f32_16x16x32_bf16(
      false, a, false, b, (short)0, c, false, false);
}

// LDS byte offset of a __shared__ pointer (generic addr low 32 bits = offset).
DEV unsigned lds_off(const void* p) { return (unsigned)(uintptr_t)p; }

// Async global->LDS 16-byte copy (per lane), tracked with ASYNCcnt.
DEV void async_b128(unsigned lds_byte, unsigned gl_byte_off, unsigned long long sbase) {
  asm volatile("global_load_async_to_lds_b128 %0, %1, %2"
               :: "v"(lds_byte), "v"(gl_byte_off), "s"(sbase) : "memory");
}
DEV void wait_async0() { asm volatile("s_wait_asynccnt 0x0" ::: "memory"); }
DEV void wait_ds0()    { asm volatile("s_wait_dscnt 0x0"    ::: "memory"); }

// A fragment: 16x32 bf16, row-major source, leading dim `ld` (elements).
// Lanes 0-15 -> M=lane, K groups {0..7,16..23}; lanes 16-31 -> {8..15,24..31}.
// Per-lane reads are two contiguous 16B chunks -> ds/global _b128.
DEV v16bf load_a(const __bf16* A, int ld, int koff) {
  const int lane = threadIdx.x & 31;
  const __bf16* p = A + (size_t)(lane & 15) * ld + koff;
  const int kh = (lane >> 4) * 8;
  v16bf a;
#pragma unroll
  for (int j = 0; j < 8; ++j) {
    a[j]     = p[kh + j];
    a[8 + j] = p[16 + kh + j];
  }
  return a;
}

// B fragment from an N-major (K-contiguous) tile T[n][k], leading dim `ld`:
// element (k, n) = T[(n0+n)*ld + k0 + k].  Lanes 0-15: K {0..7,16..23} for N=lane;
// lanes 16-31: K {8..15,24..31} for N=lane-16.  Two contiguous 16B reads per lane.
DEV v16bf load_bk(const __bf16* T, int n0, int k0, int ld) {
  const int lane = threadIdx.x & 31;
  const __bf16* p = T + (size_t)(n0 + (lane & 15)) * ld + k0;
  const int kh = (lane >> 4) * 8;
  v16bf b;
#pragma unroll
  for (int j = 0; j < 8; ++j) {
    b[j]     = p[kh + j];
    b[8 + j] = p[16 + kh + j];
  }
  return b;
}

// ---------------------------------------------------------------------------
// f32 -> bf16 conversion
// ---------------------------------------------------------------------------
__global__ void f32_to_bf16_kernel(const float* __restrict__ src,
                                   __bf16* __restrict__ dst, int n) {
  int i = blockIdx.x * blockDim.x + threadIdx.x;
  if (i < n) dst[i] = f2bf(src[i]);
}

// ---------------------------------------------------------------------------
// Tiled bf16 WMMA GEMM: C(Mx1024) = A(Mx1024) * B(1024x1024) + bias
// Block tile 128x128, BK=32, 256 threads = 8 waves (4x2), wave tile 32x64.
// A tile staged row-major via async-to-LDS; B tile staged transposed (col-
// major, K-contiguous) so B-fragment loads are two 16B LDS reads.
// ---------------------------------------------------------------------------
template <int OUT_F32>
__global__ __launch_bounds__(256)
void gemm_bf16_kernel(const __bf16* __restrict__ A, const __bf16* __restrict__ Bw,
                      const float* __restrict__ bias, void* __restrict__ Cout) {
  constexpr int N = 1024, K = 1024;
  __shared__ __align__(16) __bf16 As[128 * 32];    // row-major, ld=32
  __shared__ __align__(16) __bf16 BsT[128 * 32];   // col-major: BsT[n*32+k]

  const int tid = threadIdx.x;
  const int wave = tid >> 5, lane = tid & 31;
  const int wm = wave >> 1, wn = wave & 1;
  const int row0 = blockIdx.y * 128, col0 = blockIdx.x * 128;

  const unsigned long long a_base =
      (unsigned long long)(uintptr_t)(A + (size_t)row0 * K);

  v8f acc[2][4];
#pragma unroll
  for (int mt = 0; mt < 2; ++mt)
#pragma unroll
    for (int nt = 0; nt < 4; ++nt) acc[mt][nt] = vzero();

  for (int k0 = 0; k0 < K; k0 += 32) {
    // ---- A tile: 128x32, same-major -> async DMA straight into LDS
#pragma unroll
    for (int it = 0; it < 2; ++it) {
      const int idx = tid + it * 256;            // 512 x 16B chunks
      const int row = idx >> 2, q = idx & 3;
      async_b128(lds_off(As) + (unsigned)(row * 64 + q * 16),
                 (unsigned)((row * K + k0 + q * 8) * 2), a_base);
    }
    // ---- B tile: 32x128 row-major in memory -> transposed into BsT
#pragma unroll
    for (int it = 0; it < 2; ++it) {
      const int idx = tid + it * 256;            // 512 x 16B chunks
      const int krow = idx >> 4, q = idx & 15;   // n0 = q*8
      const uint4 raw =
          *(const uint4*)(Bw + (size_t)(k0 + krow) * N + col0 + q * 8);
      const v8bf vals = __builtin_bit_cast(v8bf, raw);
#pragma unroll
      for (int j = 0; j < 8; ++j) BsT[(q * 8 + j) * 32 + krow] = vals[j];
    }
    wait_async0();
    __syncthreads();

    v16bf af[2], bf[4];
#pragma unroll
    for (int mt = 0; mt < 2; ++mt)
      af[mt] = load_a(As + (size_t)(wm * 32 + mt * 16) * 32, 32, 0);
#pragma unroll
    for (int nt = 0; nt < 4; ++nt) bf[nt] = load_bk(BsT, wn * 64 + nt * 16, 0, 32);
#pragma unroll
    for (int mt = 0; mt < 2; ++mt)
#pragma unroll
      for (int nt = 0; nt < 4; ++nt)
        acc[mt][nt] = wmma_bf16(af[mt], bf[nt], acc[mt][nt]);
    __syncthreads();
  }

  const int mbase = (lane >> 4) * 8, nlane = lane & 15;
#pragma unroll
  for (int mt = 0; mt < 2; ++mt)
#pragma unroll
    for (int nt = 0; nt < 4; ++nt) {
      const int row = row0 + wm * 32 + mt * 16 + mbase;
      const int col = col0 + wn * 64 + nt * 16 + nlane;
      const float bv = bias[col];
#pragma unroll
      for (int r = 0; r < 8; ++r) {
        const size_t idx = ((size_t)(row + r) << 10) + col;   // N = 1024
        const float v = acc[mt][nt][r] + bv;
        if (OUT_F32) ((float*)Cout)[idx] = v;
        else         ((__bf16*)Cout)[idx] = f2bf(v);
      }
    }
}

// ---------------------------------------------------------------------------
// Flash attention with dual scores.
// Grid: (S/64, B*H).  Block: 128 threads = 4 waves; wave w owns 16 q-rows.
// K_sem/K_sal tiles (key-major, consumed K-contiguously) staged via async
// DMA; V tile staged transposed (dk-major) for contiguous B-fragment loads.
// ---------------------------------------------------------------------------
__global__ __launch_bounds__(128)
void attn_kernel(const __bf16* __restrict__ Qsem, const __bf16* __restrict__ Ksem,
                 const __bf16* __restrict__ Vb,   const __bf16* __restrict__ Qsal,
                 const __bf16* __restrict__ Ksal, const int* __restrict__ mask,
                 const float* __restrict__ gammap, __bf16* __restrict__ Obuf) {
  __shared__ __align__(16) __bf16 KsS[32 * 64];   // K_sem tile, key-major
  __shared__ __align__(16) __bf16 KlS[32 * 64];   // K_sal tile, key-major
  __shared__ __align__(16) __bf16 Vt[64 * 32];    // V tile, dk-major: Vt[dk*32+key]
  __shared__ __align__(16) __bf16 Ps[4][16 * 32]; // per-wave P tile, row-major

  const int bh = blockIdx.y, b = bh / Hh, h = bh % Hh;
  const int wave = threadIdx.x >> 5, lane = threadIdx.x & 31;
  const int q0 = blockIdx.x * 64 + wave * 16;
  const float scale = 0.125f;                     // 1/sqrt(64)
  const float g = *gammap;

  const size_t head_off = (size_t)b * Ss * Dd + h * DKk;
  const unsigned long long ks_base = (unsigned long long)(uintptr_t)(Ksem + head_off);
  const unsigned long long kl_base = (unsigned long long)(uintptr_t)(Ksal + head_off);

  const size_t qoff = head_off + (size_t)q0 * Dd;
  const v16bf qs0 = load_a(Qsem + qoff, Dd, 0);
  const v16bf qs1 = load_a(Qsem + qoff, Dd, 32);
  const v16bf ql0 = load_a(Qsal + qoff, Dd, 0);
  const v16bf ql1 = load_a(Qsal + qoff, Dd, 32);

  float run_m[8], run_l[8];
  v8f Oacc[4];
#pragma unroll
  for (int r = 0; r < 8; ++r) { run_m[r] = -3.0e38f; run_l[r] = 0.0f; }
#pragma unroll
  for (int dt = 0; dt < 4; ++dt) Oacc[dt] = vzero();

  const int mbase = (lane >> 4) * 8, nlane = lane & 15;

  for (int k0 = 0; k0 < Ss; k0 += 32) {
    {  // ---- stage tiles
      const int row = threadIdx.x >> 2, quad = threadIdx.x & 3;  // row: key 0..31
      const unsigned gb0 = (unsigned)(((k0 + row) * Dd + quad * 8) * 2);
      const unsigned lb0 = (unsigned)(row * 128 + quad * 16);
      // K_sem / K_sal: same-major -> async DMA (2x16B each per thread)
      async_b128(lds_off(KsS) + lb0,      gb0,      ks_base);
      async_b128(lds_off(KsS) + lb0 + 64, gb0 + 64, ks_base);
      async_b128(lds_off(KlS) + lb0,      gb0,      kl_base);
      async_b128(lds_off(KlS) + lb0 + 64, gb0 + 64, kl_base);
      // V: transpose into dk-major LDS
#pragma unroll
      for (int half = 0; half < 2; ++half) {
        const int dk0 = quad * 8 + half * 32;
        const uint4 raw =
            *(const uint4*)(Vb + head_off + (size_t)(k0 + row) * Dd + dk0);
        const v8bf vals = __builtin_bit_cast(v8bf, raw);
#pragma unroll
        for (int j = 0; j < 8; ++j) Vt[(dk0 + j) * 32 + row] = vals[j];
      }
    }
    wait_async0();
    __syncthreads();

    // ---- scores for 32 keys: two 16x16 D tiles per wave
    v8f sc[2];
#pragma unroll
    for (int nt = 0; nt < 2; ++nt) {
      v8f t = vzero();
      t = wmma_bf16(ql0, load_bk(KlS, nt * 16, 0, 64), t);
      t = wmma_bf16(ql1, load_bk(KlS, nt * 16, 32, 64), t);
      v8f s = vzero();
      s = wmma_bf16(qs0, load_bk(KsS, nt * 16, 0, 64), s);
      s = wmma_bf16(qs1, load_bk(KsS, nt * 16, 32, 64), s);
      const int key = k0 + nt * 16 + nlane;
#pragma unroll
      for (int r = 0; r < 8; ++r) {
        float v = scale * (s[r] + g * t[r]);
        const int qrow = q0 + mbase + r;
        if (mask[(size_t)(b * Ss + qrow) * Ss + key] == 0) v = -1.0e9f;
        sc[nt][r] = v;
      }
    }

    // ---- online softmax: rows in VGPR index r, keys in lanes (per half-wave)
#pragma unroll
    for (int r = 0; r < 8; ++r) {
      float rm = fmaxf(sc[0][r], sc[1][r]);
      rm = fmaxf(rm, __shfl_xor(rm, 1));
      rm = fmaxf(rm, __shfl_xor(rm, 2));
      rm = fmaxf(rm, __shfl_xor(rm, 4));
      rm = fmaxf(rm, __shfl_xor(rm, 8));
      const float nm = fmaxf(run_m[r], rm);
      const float corr = __expf(run_m[r] - nm);
      const float p0 = __expf(sc[0][r] - nm);
      const float p1 = __expf(sc[1][r] - nm);
      float ps = p0 + p1;
      ps += __shfl_xor(ps, 1);
      ps += __shfl_xor(ps, 2);
      ps += __shfl_xor(ps, 4);
      ps += __shfl_xor(ps, 8);
      run_l[r] = run_l[r] * corr + ps;
      run_m[r] = nm;
#pragma unroll
      for (int dt = 0; dt < 4; ++dt) Oacc[dt][r] *= corr;
      Ps[wave][(mbase + r) * 32 + nlane]      = f2bf(p0);
      Ps[wave][(mbase + r) * 32 + 16 + nlane] = f2bf(p1);
    }

    // ---- O += P * V  (P store -> A-fragment load round trip via LDS)
    wait_ds0();
    const v16bf pf = load_a(Ps[wave], 32, 0);
#pragma unroll
    for (int dt = 0; dt < 4; ++dt)
      Oacc[dt] = wmma_bf16(pf, load_bk(Vt, dt * 16, 0, 32), Oacc[dt]);
    __syncthreads();
  }

  // ---- normalize and store O as bf16 in (B,S,D) layout for the Wo GEMM
#pragma unroll
  for (int dt = 0; dt < 4; ++dt)
#pragma unroll
    for (int r = 0; r < 8; ++r) {
      const int qrow = q0 + mbase + r;
      const int dk = dt * 16 + nlane;
      const float o = Oacc[dt][r] / run_l[r];
      Obuf[head_off + (size_t)qrow * Dd + dk] = f2bf(o);
    }
}

// ---------------------------------------------------------------------------
// Launch
// ---------------------------------------------------------------------------
extern "C" void kernel_launch(void* const* d_in, const int* in_sizes, int n_in,
                              void* d_out, int out_size, void* d_ws, size_t ws_size,
                              hipStream_t stream) {
  (void)in_sizes; (void)n_in; (void)out_size; (void)ws_size;
  const float* X_sem  = (const float*)d_in[0];
  const float* X_sal  = (const float*)d_in[1];
  const int*   mask   = (const int*)d_in[2];
  const float* Wq_sem = (const float*)d_in[3];
  const float* bq_sem = (const float*)d_in[4];
  const float* Wk_sem = (const float*)d_in[5];
  const float* bk_sem = (const float*)d_in[6];
  const float* Wv     = (const float*)d_in[7];
  const float* bv     = (const float*)d_in[8];
  const float* Wq_sal = (const float*)d_in[9];
  const float* bq_sal = (const float*)d_in[10];
  const float* Wk_sal = (const float*)d_in[11];
  const float* bk_sal = (const float*)d_in[12];
  const float* Wo     = (const float*)d_in[13];
  const float* bo     = (const float*)d_in[14];
  const float* gamma  = (const float*)d_in[15];

  char* ws = (char*)d_ws;
  size_t off = 0;
  auto carve = [&](size_t elems) {
    __bf16* p = (__bf16*)(ws + off);
    off += ((elems * 2 + 255) & ~(size_t)255);
    return p;
  };
  const size_t ND = (size_t)Bb * Ss * Dd;  // 4M activations
  const size_t WD = (size_t)Dd * Dd;       // 1M weights

  __bf16* xsem = carve(ND);  __bf16* xsal = carve(ND);
  __bf16* wqs = carve(WD);   __bf16* wks = carve(WD);  __bf16* wv_ = carve(WD);
  __bf16* wql = carve(WD);   __bf16* wkl = carve(WD);  __bf16* wo_ = carve(WD);
  __bf16* Qs = carve(ND);    __bf16* Kse = carve(ND);  __bf16* Vv = carve(ND);
  __bf16* Ql = carve(ND);    __bf16* Kla = carve(ND);  __bf16* Ob = carve(ND);

  const int cb = 256;
  f32_to_bf16_kernel<<<(int)((ND + cb - 1) / cb), cb, 0, stream>>>(X_sem, xsem, (int)ND);
  f32_to_bf16_kernel<<<(int)((ND + cb - 1) / cb), cb, 0, stream>>>(X_sal, xsal, (int)ND);
  f32_to_bf16_kernel<<<(int)((WD + cb - 1) / cb), cb, 0, stream>>>(Wq_sem, wqs, (int)WD);
  f32_to_bf16_kernel<<<(int)((WD + cb - 1) / cb), cb, 0, stream>>>(Wk_sem, wks, (int)WD);
  f32_to_bf16_kernel<<<(int)((WD + cb - 1) / cb), cb, 0, stream>>>(Wv,     wv_, (int)WD);
  f32_to_bf16_kernel<<<(int)((WD + cb - 1) / cb), cb, 0, stream>>>(Wq_sal, wql, (int)WD);
  f32_to_bf16_kernel<<<(int)((WD + cb - 1) / cb), cb, 0, stream>>>(Wk_sal, wkl, (int)WD);
  f32_to_bf16_kernel<<<(int)((WD + cb - 1) / cb), cb, 0, stream>>>(Wo,     wo_, (int)WD);

  dim3 gg(Dd / 128, (Bb * Ss) / 128);  // (8, 32)
  gemm_bf16_kernel<0><<<gg, 256, 0, stream>>>(xsem, wqs, bq_sem, Qs);
  gemm_bf16_kernel<0><<<gg, 256, 0, stream>>>(xsem, wks, bk_sem, Kse);
  gemm_bf16_kernel<0><<<gg, 256, 0, stream>>>(xsem, wv_, bv,     Vv);
  gemm_bf16_kernel<0><<<gg, 256, 0, stream>>>(xsal, wql, bq_sal, Ql);
  gemm_bf16_kernel<0><<<gg, 256, 0, stream>>>(xsal, wkl, bk_sal, Kla);

  dim3 ag(Ss / 64, Bb * Hh);           // (32, 32)
  attn_kernel<<<ag, 128, 0, stream>>>(Qs, Kse, Vv, Ql, Kla, mask, gamma, Ob);

  gemm_bf16_kernel<1><<<gg, 256, 0, stream>>>(Ob, wo_, bo, d_out);
}
